// Module_GAT_LSTM_split_49254684950645
// MI455X (gfx1250) — matrix operand order repaired and checked by reading, not compiled, not run on hardware
//
#include <hip/hip_runtime.h>

// ---------------------------------------------------------------------------
// Problem constants (from reference)
// ---------------------------------------------------------------------------
#define BB     8
#define NN     256
#define BN     2048          // B*N rows
#define D_IN   7
#define D_H    256
#define D_KQV  128
#define NHEAD  6
#define MASK_C 10000.0f

typedef __attribute__((ext_vector_type(16))) _Float16 v16h;
typedef __attribute__((ext_vector_type(8)))  _Float16 v8h;
typedef __attribute__((ext_vector_type(8)))  float    v8f;

#define ACT_NONE 0
#define ACT_MISH 1

// ---------------------------------------------------------------------------
// Fast activation helpers (saturating closed forms over __expf)
// ---------------------------------------------------------------------------
__device__ __forceinline__ float sigmoid_f(float x) {
  return __fdividef(1.0f, 1.0f + __expf(-x));
}
__device__ __forceinline__ float fast_tanh(float x) {
  float cx = fminf(fmaxf(x, -15.0f), 15.0f);
  float t  = __expf(2.0f * cx);
  return __fdividef(t - 1.0f, t + 1.0f);
}
// mish(x) = x * tanh(log1p(e^x)) = x * (t^2 + 2t) / (t^2 + 2t + 2),  t = e^x
__device__ __forceinline__ float mish_f(float x) {
  float t = __expf(fminf(x, 20.0f));          // t^2 <= 2.4e17, no overflow
  float u = t * t + 2.0f * t;
  float r = x * __fdividef(u, u + 2.0f);
  return (x > 20.0f) ? x : r;
}

// ---------------------------------------------------------------------------
// WMMA core: 16xK A-panel  x  (NB*16)-wide Bt panel -> NB accumulators.
// A : [M,K] row-major f16.   Bt: [N,K] row-major f16 (i.e. B transposed),
// so each lane's fragments are contiguous 128/256-bit loads.
// Fragment layouts per CDNA5 ISA 7.12.2 (wave32).
// ---------------------------------------------------------------------------
template<int NB>
__device__ __forceinline__ void wmma_tiles(const _Float16* __restrict__ A, int lda,
                                           const _Float16* __restrict__ Bt, int ldbt,
                                           int m0, int n0, int K, v8f* acc)
{
  const int lane = threadIdx.x & 31;
  const int half = lane >> 4;
  const int lm   = lane & 15;
  const _Float16* Arow = A + (size_t)(m0 + lm) * lda;
  const _Float16* Brow[NB];
#pragma unroll
  for (int t = 0; t < NB; ++t)
    Brow[t] = Bt + (size_t)(n0 + 16 * t + lm) * ldbt;
  const int ka0 = half ? 8 : 0;       // A: lanes 0-15 hold K 0..7 /16..23
  const int ka1 = half ? 24 : 16;     //    lanes16-31 hold K 8..15/24..31
                                      // B: same K order as A
  for (int kk = 0; kk < K; kk += 32) {
    v8h a0 = *(const v8h*)(Arow + kk + ka0);
    v8h a1 = *(const v8h*)(Arow + kk + ka1);
    v16h af = __builtin_shufflevector(a0, a1, 0, 1, 2, 3, 4, 5, 6, 7,
                                      8, 9, 10, 11, 12, 13, 14, 15);
#pragma unroll
    for (int t = 0; t < NB; ++t) {
      v8h b0 = *(const v8h*)(Brow[t] + kk + ka0);
      v8h b1 = *(const v8h*)(Brow[t] + kk + ka1);
      v16h bf = __builtin_shufflevector(b0, b1, 0, 1, 2, 3, 4, 5, 6, 7,
                                        8, 9, 10, 11, 12, 13, 14, 15);
      acc[t] = __builtin_amdgcn_wmma_f32_16x16x32_f16(false, af, false, bf,
                                                      (short)0, acc[t], false, false);
    }
  }
}

// ---------------------------------------------------------------------------
// Generic WMMA GEMM: C = act(A@B + Cin + bias).
// Tile: 16 (M) x 64 (N) per wave.  grid = (N/64, M/16), block = 32.
// ---------------------------------------------------------------------------
template<int ACT, bool CIN, bool BIAS, bool O32, bool O16, bool O16T>
__global__ void gemm_k(const _Float16* __restrict__ A, int lda,
                       const _Float16* __restrict__ Bt, int ldbt,
                       const float* __restrict__ bias,
                       const float* __restrict__ Cin, int ldci,
                       float* __restrict__ C32, int ldc32,
                       _Float16* __restrict__ C16, int ldc16,
                       _Float16* __restrict__ C16T, int ldc16t,
                       int K)
{
  const int n0 = blockIdx.x * 64;
  const int m0 = blockIdx.y * 16;
  v8f acc[4] = {};
  wmma_tiles<4>(A, lda, Bt, ldbt, m0, n0, K, acc);
  const int lane = threadIdx.x & 31;
  const int half = lane >> 4;
#pragma unroll
  for (int t = 0; t < 4; ++t) {
    const int col = n0 + 16 * t + (lane & 15);
#pragma unroll
    for (int r = 0; r < 8; ++r) {
      const int row = m0 + r + 8 * half;   // C/D layout: VGPR r -> M=r (+8 hi lanes)
      float v = acc[t][r];
      if (CIN)  v += Cin[(size_t)row * ldci + col];
      if (BIAS) v += bias[col];
      if (ACT == ACT_MISH) v = mish_f(v);
      if (O32)  C32[(size_t)row * ldc32 + col] = v;
      if (O16)  C16[(size_t)row * ldc16 + col] = (_Float16)v;
      if (O16T) C16T[(size_t)col * ldc16t + row] = (_Float16)v;
    }
  }
}

// ---------------------------------------------------------------------------
// Batched attention aggregation: z = b*6+h,
//   p_att[b,:,h,:] = w[b,h][256x256] @ v[b][256x128]
// B operand taken from transposed vT[128][2048] at column offset b*256.
// Output written into decoder input din[h][2048][256] at column offset 128.
// grid = (128/64, 256/16, 48), block = 32.
// ---------------------------------------------------------------------------
__global__ void gemm_patt(const _Float16* __restrict__ wAll,
                          const _Float16* __restrict__ vT,
                          _Float16* __restrict__ din)
{
  const int z = blockIdx.z;
  const int b = z / NHEAD, h = z % NHEAD;
  const _Float16* A  = wAll + (size_t)z * (NN * NN);
  const _Float16* Bt = vT + (size_t)b * NN;                     // ldbt = BN
  _Float16* C = din + (size_t)h * (BN * 256) + (size_t)b * (NN * 256) + D_KQV;
  const int n0 = blockIdx.x * 64;
  const int m0 = blockIdx.y * 16;
  v8f acc[4] = {};
  wmma_tiles<4>(A, NN, Bt, BN, m0, n0, NN, acc);
  const int lane = threadIdx.x & 31;
  const int half = lane >> 4;
#pragma unroll
  for (int t = 0; t < 4; ++t) {
    const int col = n0 + 16 * t + (lane & 15);
#pragma unroll
    for (int r = 0; r < 8; ++r) {
      const int row = m0 + r + 8 * half;
      C[(size_t)row * 256 + col] = (_Float16)acc[t][r];
    }
  }
}

// ---------------------------------------------------------------------------
// Elementwise / small kernels
// ---------------------------------------------------------------------------
__global__ void cvt_f16(const float* __restrict__ s, _Float16* __restrict__ d, int n)
{
  int i = blockIdx.x * blockDim.x + threadIdx.x;
  if (i < n) d[i] = (_Float16)s[i];
}

// src [rows, cols] f32 -> dst [cols, rows] f16   (weight -> Bt layout)
__global__ void cvtT_f16(const float* __restrict__ s, _Float16* __restrict__ d,
                         int rows, int cols)
{
  int i = blockIdx.x * blockDim.x + threadIdx.x;
  if (i < rows * cols) {
    int r = i / cols, c = i % cols;
    d[(size_t)c * rows + r] = (_Float16)s[i];
  }
}

// pre[m,n] = sum_k x[m,k] * W_ih[n,k] + b_ih[n] + b_hh[n]      (K = 7, scalar)
__global__ void lstm_pre(const float* __restrict__ x, const float* __restrict__ Wih,
                         const float* __restrict__ bih, const float* __restrict__ bhh,
                         float* __restrict__ pre)
{
  int idx = blockIdx.x * blockDim.x + threadIdx.x;   // m*1024 + n
  int n = idx & 1023, m = idx >> 10;
  const float* xr = x + m * D_IN;
  const float* wr = Wih + n * D_IN;
  float s = bih[n] + bhh[n];
#pragma unroll
  for (int k = 0; k < D_IN; ++k) s += xr[k] * wr[k];
  pre[idx] = s;
}

// gates (i,f,g,o) -> h_new, c_new; also f16 copy of h for downstream GEMMs
__global__ void lstm_gates(const float* __restrict__ G, const float* __restrict__ cell,
                           float* __restrict__ h_out, float* __restrict__ c_out,
                           _Float16* __restrict__ h16)
{
  int idx = blockIdx.x * blockDim.x + threadIdx.x;   // m*256 + j
  int j = idx & 255, m = idx >> 8;
  const float* g = G + (size_t)m * 1024;
  float i_ = sigmoid_f(g[j]);
  float f_ = sigmoid_f(g[256 + j]);
  float gg = fast_tanh(g[512 + j]);
  float o_ = sigmoid_f(g[768 + j]);
  float cn = f_ * cell[idx] + i_ * gg;
  float hn = o_ * fast_tanh(cn);
  h_out[idx] = hn;
  c_out[idx] = cn;
  h16[idx]   = (_Float16)hn;
}

// w[b,h,i,j] = sigmoid( tanh(Aatt_i + Bq_j) @ W2[:,h] + b2[h] - mask )  -> f16
__global__ void pairwise(const float* __restrict__ Aatt, const float* __restrict__ Bq,
                         const float* __restrict__ W2, const float* __restrict__ b2,
                         const float* __restrict__ x, _Float16* __restrict__ w16)
{
  __shared__ float w2s[D_KQV * NHEAD];
  __shared__ float b2s[NHEAD];
  for (int t = threadIdx.x; t < D_KQV * NHEAD; t += blockDim.x) w2s[t] = W2[t];
  if (threadIdx.x < NHEAD) b2s[threadIdx.x] = b2[threadIdx.x];
  __syncthreads();

  int idx = blockIdx.x * blockDim.x + threadIdx.x;   // b*65536 + i*256 + j
  int j = idx & 255, i = (idx >> 8) & 255, b = idx >> 16;
  bool dead_i = x[(size_t)(b * NN + i) * D_IN + (D_IN - 1)] != 1.0f;
  bool dead_j = x[(size_t)(b * NN + j) * D_IN + (D_IN - 1)] != 1.0f;
  float m = (dead_i || dead_j || (i == j)) ? MASK_C : 0.0f;
  const float* a = Aatt + (size_t)(b * NN + i) * D_KQV;
  const float* q = Bq   + (size_t)(b * NN + j) * D_KQV;
  float acc[NHEAD] = {};
  for (int t = 0; t < D_KQV; ++t) {
    float hv = fast_tanh(a[t] + q[t]);
#pragma unroll
    for (int h = 0; h < NHEAD; ++h) acc[h] += hv * w2s[t * NHEAD + h];
  }
#pragma unroll
  for (int h = 0; h < NHEAD; ++h) {
    float wv = sigmoid_f(acc[h] + b2s[h] - m);
    w16[((size_t)(b * NHEAD + h) * NN + i) * NN + j] = (_Float16)wv;
  }
}

// din[h][m][0:128] = v[m]   (columns 128:256 filled by gemm_patt)
__global__ void fill_din_v(const _Float16* __restrict__ v16, _Float16* __restrict__ din)
{
  int i = blockIdx.x * blockDim.x + threadIdx.x;   // h*(2048*128) + m*128 + d
  int d = i & 127, m = (i >> 7) & 2047, h = i >> 18;
  din[(size_t)h * (BN * 256) + (size_t)m * 256 + d] = v16[(size_t)m * D_KQV + d];
}

// final 64->1 heads: mu = t_mu@Wm2 + bm2 ; sig = sigmoid(t_sig@Ws2 + bs2) + 1e-6
__global__ void head_out(const _Float16* __restrict__ tmu, const _Float16* __restrict__ tsg,
                         const float* __restrict__ Wm2, const float* __restrict__ bm2,
                         const float* __restrict__ Ws2, const float* __restrict__ bs2,
                         float* __restrict__ omu, float* __restrict__ osg)
{
  int m = blockIdx.x * blockDim.x + threadIdx.x;   // 0..2047
  const _Float16* a = tmu + (size_t)m * 64;
  const _Float16* s = tsg + (size_t)m * 64;
  float smu = bm2[0], ssg = bs2[0];
  for (int k = 0; k < 64; ++k) {
    smu += (float)a[k] * Wm2[k];
    ssg += (float)s[k] * Ws2[k];
  }
  omu[m] = smu;                               // MU_NORM == 1
  osg[m] = sigmoid_f(ssg) + 1e-6f;            // SIG_NORM == 1
}

// ---------------------------------------------------------------------------
// Host launcher
// ---------------------------------------------------------------------------
extern "C" void kernel_launch(void* const* d_in, const int* in_sizes, int n_in,
                              void* d_out, int out_size, void* d_ws, size_t ws_size,
                              hipStream_t stream)
{
  (void)n_in; (void)out_size; (void)ws_size;
  auto F = [&](int i) -> const float* { return (const float*)d_in[i]; };

  const float* x      = F(0);
  const float* hidden = F(1);
  const float* cell   = F(2);
  // lstm_params dict: disambiguate flatten order via sizes (W_hh=262144, W_ih=7168)
  const float *W_hh, *W_ih;
  if (in_sizes[3] == 4 * D_H * D_H) { W_hh = F(3); W_ih = F(4); }
  else                              { W_ih = F(3); W_hh = F(4); }
  const float* b_hh = F(5);
  const float* b_ih = F(6);   // order irrelevant: only b_ih + b_hh is used
  const int KQV0 = 7;  // k:7..10, q:11..14, v:15..18 ; att:19..22 ; dec/mu/sig follow

  float* out   = (float*)d_out;
  float* h_out = out + 2 * NHEAD * BN;        // after 12 x [B,N]
  float* c_out = h_out + BN * D_H;

  // ---- workspace carve-up ----
  char*  wsp = (char*)d_ws;
  size_t cur = 0;
  auto allocF = [&](size_t n) -> float* {
    cur = (cur + 255) & ~(size_t)255; float* p = (float*)(wsp + cur); cur += n * 4; return p;
  };
  auto allocH = [&](size_t n) -> _Float16* {
    cur = (cur + 255) & ~(size_t)255; _Float16* p = (_Float16*)(wsp + cur); cur += n * 2; return p;
  };

  float*    G     = allocF((size_t)BN * 1024);     // LSTM pre-activations / gates
  float*    Aatt  = allocF((size_t)BN * D_KQV);
  float*    Bqv   = allocF((size_t)BN * D_KQV);
  _Float16* hIn16 = allocH((size_t)BN * D_H);      // input hidden, f16
  _Float16* WhhBt = allocH((size_t)(4 * D_H) * D_H); // W_hh flat = Bt layout [1024][256]
  _Float16 *Wl1t[3], *Wl2t[3];
  for (int p = 0; p < 3; ++p) { Wl1t[p] = allocH(256 * 256); Wl2t[p] = allocH(256 * 128); }
  _Float16* aW1t  = allocH(256 * 128);             // att W1^T [128][256]
  _Float16 *dW1t[NHEAD], *dW2t[NHEAD], *mW1t[NHEAD], *sW1t[NHEAD];
  for (int h = 0; h < NHEAD; ++h) {
    dW1t[h] = allocH(256 * 256); dW2t[h] = allocH(256 * 128);
    mW1t[h] = allocH(128 * 64);  sW1t[h] = allocH(128 * 64);
  }
  _Float16* h16  = allocH((size_t)BN * D_H);       // new hidden, f16
  _Float16* t1   = allocH((size_t)BN * 256);       // k/q/v hidden (reused)
  _Float16* kqv16[3];
  for (int p = 0; p < 3; ++p) kqv16[p] = allocH((size_t)BN * D_KQV);
  _Float16* vT   = allocH((size_t)D_KQV * BN);     // v transposed [128][2048]
  _Float16* w16  = allocH((size_t)BB * NHEAD * NN * NN);   // attention weights
  _Float16* din  = allocH((size_t)NHEAD * BN * 256);       // [v | p_att_h] per head
  _Float16* t2   = allocH((size_t)BN * 256);       // decoder hidden (reused)
  _Float16* d16  = allocH((size_t)BN * 128);       // decoder output (reused)
  _Float16* tmu  = allocH((size_t)BN * 64);
  _Float16* tsg  = allocH((size_t)BN * 64);

  auto cvt = [&](const float* s, _Float16* d, int n) {
    cvt_f16<<<(n + 255) / 256, 256, 0, stream>>>(s, d, n);
  };
  auto cvtT = [&](const float* s, _Float16* d, int rows, int cols) {
    cvtT_f16<<<(rows * cols + 255) / 256, 256, 0, stream>>>(s, d, rows, cols);
  };

  // ---- weight / input conversions (all weights into transposed Bt layout) ----
  cvt(hidden, hIn16, BN * D_H);
  cvt(W_hh, WhhBt, 4 * D_H * D_H);                 // [N=1024][K=256] already
  for (int p = 0; p < 3; ++p) {
    cvtT(F(KQV0 + 4 * p + 0), Wl1t[p], 256, 256);
    cvtT(F(KQV0 + 4 * p + 2), Wl2t[p], 256, 128);
  }
  cvtT(F(19), aW1t, 256, 128);                     // -> [128][256]
  for (int h = 0; h < NHEAD; ++h) {
    cvtT(F(23 + 4 * h + 0), dW1t[h], 256, 256);
    cvtT(F(23 + 4 * h + 2), dW2t[h], 256, 128);
    cvtT(F(47 + 4 * h + 0), mW1t[h], 128, 64);
    cvtT(F(71 + 4 * h + 0), sW1t[h], 128, 64);
  }

  // ---- LSTM ----
  lstm_pre<<<(BN * 1024) / 256, 256, 0, stream>>>(x, W_ih, b_ih, b_hh, G);
  gemm_k<ACT_NONE, true, false, true, false, false>
      <<<dim3(1024 / 64, BN / 16), 32, 0, stream>>>(
      hIn16, D_H, WhhBt, D_H, nullptr, G, 1024, G, 1024, nullptr, 0, nullptr, 0, D_H);
  lstm_gates<<<(BN * D_H) / 256, 256, 0, stream>>>(G, cell, h_out, c_out, h16);

  // ---- k / q / v MLPs (mish hidden, linear out) ----
  for (int p = 0; p < 3; ++p) {
    gemm_k<ACT_MISH, false, true, false, true, false>
        <<<dim3(256 / 64, BN / 16), 32, 0, stream>>>(
        h16, D_H, Wl1t[p], 256, F(KQV0 + 4 * p + 1), nullptr, 0,
        nullptr, 0, t1, 256, nullptr, 0, D_H);
    if (p == 2) {  // v: also produce transposed copy for the aggregation GEMMs
      gemm_k<ACT_NONE, false, true, false, true, true>
          <<<dim3(128 / 64, BN / 16), 32, 0, stream>>>(
          t1, 256, Wl2t[p], 256, F(KQV0 + 4 * p + 3), nullptr, 0,
          nullptr, 0, kqv16[p], 128, vT, BN, 256);
    } else {
      gemm_k<ACT_NONE, false, true, false, true, false>
          <<<dim3(128 / 64, BN / 16), 32, 0, stream>>>(
          t1, 256, Wl2t[p], 256, F(KQV0 + 4 * p + 3), nullptr, 0,
          nullptr, 0, kqv16[p], 128, nullptr, 0, 256);
    }
  }

  // ---- attention layer-1, factorized over the concat ----
  gemm_k<ACT_NONE, false, true, true, false, false>
      <<<dim3(128 / 64, BN / 16), 32, 0, stream>>>(
      kqv16[0], 128, aW1t, 256, F(20), nullptr, 0,
      Aatt, 128, nullptr, 0, nullptr, 0, 128);               // k-part (+b1)
  gemm_k<ACT_NONE, false, false, true, false, false>
      <<<dim3(128 / 64, BN / 16), 32, 0, stream>>>(
      kqv16[1], 128, aW1t + 128, 256, nullptr, nullptr, 0,
      Bqv, 128, nullptr, 0, nullptr, 0, 128);                // q-part

  // ---- pairwise tanh + 128->6 + mask + sigmoid ----
  pairwise<<<(BB * NN * NN) / 256, 256, 0, stream>>>(Aatt, Bqv, F(21), F(22), x, w16);

  // ---- decoder inputs: [v | w@v] ----
  fill_din_v<<<(NHEAD * BN * D_KQV) / 256, 256, 0, stream>>>(kqv16[2], din);
  gemm_patt<<<dim3(128 / 64, NN / 16, BB * NHEAD), 32, 0, stream>>>(w16, vT, din);

  // ---- per-head decoder + mu/sig heads ----
  for (int h = 0; h < NHEAD; ++h) {
    const _Float16* dinh = din + (size_t)h * (BN * 256);
    gemm_k<ACT_MISH, false, true, false, true, false>
        <<<dim3(256 / 64, BN / 16), 32, 0, stream>>>(
        dinh, 256, dW1t[h], 256, F(24 + 4 * h), nullptr, 0,
        nullptr, 0, t2, 256, nullptr, 0, 256);
    gemm_k<ACT_NONE, false, true, false, true, false>
        <<<dim3(128 / 64, BN / 16), 32, 0, stream>>>(
        t2, 256, dW2t[h], 256, F(26 + 4 * h), nullptr, 0,
        nullptr, 0, d16, 128, nullptr, 0, 256);
    gemm_k<ACT_MISH, false, true, false, true, false>
        <<<dim3(64 / 64, BN / 16), 32, 0, stream>>>(
        d16, 128, mW1t[h], 128, F(48 + 4 * h), nullptr, 0,
        nullptr, 0, tmu, 64, nullptr, 0, 128);
    gemm_k<ACT_MISH, false, true, false, true, false>
        <<<dim3(64 / 64, BN / 16), 32, 0, stream>>>(
        d16, 128, sW1t[h], 128, F(72 + 4 * h), nullptr, 0,
        nullptr, 0, tsg, 64, nullptr, 0, 128);
    head_out<<<BN / 256, 256, 0, stream>>>(tmu, tsg,
        F(49 + 4 * h), F(50 + 4 * h), F(73 + 4 * h), F(74 + 4 * h),
        out + (size_t)(2 * h) * BN, out + (size_t)(2 * h + 1) * BN);
  }
}
